// AttentionModel_83502754169096
// MI455X (gfx1250) — compile-verified, matrix operand order checked by reading, not run on hardware
//
#include <hip/hip_runtime.h>
#include <math.h>

// ---------------- CDNA5 WMMA types ----------------
typedef __attribute__((ext_vector_type(16))) __bf16 v16bf;
typedef __attribute__((ext_vector_type(8)))  float  v8f;
typedef int v4i_ls __attribute__((vector_size(16)));

union FragU { v16bf v; unsigned u[8]; };

#define BB   512
#define NN   100
#define DD   128
#define HH   8
#define DK   16
#define FF_  512
#define MROWS (BB*NN)          // 51200
#define NEGI (-1e9f)

#if __has_builtin(__builtin_amdgcn_global_load_async_to_lds_b128)
#define HAS_ASYNC_LDS 1
#endif

__device__ __forceinline__ void copy16_g2l(const __bf16* g, __bf16* l) {
#ifdef HAS_ASYNC_LDS
  __builtin_amdgcn_global_load_async_to_lds_b128(
      (__attribute__((address_space(1))) v4i_ls*)g,
      (__attribute__((address_space(3))) v4i_ls*)l, 0, 0);
#else
  *(uint4*)l = *(const uint4*)g;
#endif
}

__device__ __forceinline__ void async_wait0() {
#ifdef HAS_ASYNC_LDS
#if __has_builtin(__builtin_amdgcn_s_wait_asynccnt)
  __builtin_amdgcn_s_wait_asynccnt(0);
#else
  asm volatile("s_wait_asynccnt 0x0" ::: "memory");
#endif
#endif
}

// K-pair index mapping for 16-bit A/B fragments (16x32 / 32x16), wave32:
// pair p (0..7), lane-half h: p<4 -> k = 8h + 2p ; p>=4 -> k = 16 + 8h + 2(p-4)
__device__ __forceinline__ int frag_k(int p, int half) {
  return (p < 4) ? (half * 8 + 2 * p) : (16 + half * 8 + 2 * (p - 4));
}

// ---------------- weight conversion f32 -> bf16 ----------------
__global__ void k_f2bf(const float* __restrict__ s, __bf16* __restrict__ d, int n) {
  int i = blockIdx.x * 256 + threadIdx.x;
  if (i < n) d[i] = (__bf16)s[i];
}

// ---------------- init projection: h = x @ init_W + b ----------------
__global__ void k_init(const float* __restrict__ x, const float* __restrict__ W,
                       const float* __restrict__ b, float* __restrict__ h,
                       __bf16* __restrict__ hb) {
  size_t i = (size_t)blockIdx.x * 256 + threadIdx.x;   // over MROWS*DD
  int d = (int)(i & (DD - 1));
  size_t row = i >> 7;
  float v = x[row * 2] * W[d] + x[row * 2 + 1] * W[DD + d] + b[d];
  h[i] = v;
  hb[i] = (__bf16)v;
}

// ---------------- bf16 WMMA GEMM, 64x64 block tile, 4 waves, double-buffered ----
// C[M,N] = A[M,K] @ W[K,N];  flags: bit0 bias, bit1 relu, bit2 write bf16
// grid = (N/64, M/64), block = 128 threads (4 waves); wave w owns row-tile w
// and computes 4 col-tiles (1 A-frag reused against 4 B-frags).
// A tile staged with GLOBAL_LOAD_ASYNC_TO_LDS_B128 (ASYNCcnt), ping-pong LDS.
__global__ void k_gemm(const __bf16* __restrict__ A, const __bf16* __restrict__ W,
                       const float* __restrict__ bias, float* __restrict__ Cf,
                       __bf16* __restrict__ Cb, int M, int K, int N, int flags) {
  __shared__ __align__(16) __bf16 lA[2][64 * 32];   // row-major [m][k]
  __shared__ __align__(16) __bf16 lBt[2][64 * 32];  // transposed [n][k]
  const int t = threadIdx.x;
  const int lane = t & 31, wr = t >> 5;             // wave row-tile
  const int half = lane >> 4, lid = lane & 15;
  const int m0 = blockIdx.y * 64, n0 = blockIdx.x * 64;

  auto stageA = [&](int buf, int k0) {
#pragma unroll
    for (int idx = t; idx < 256; idx += 128) {
      int r = idx >> 2, q = idx & 3;
      copy16_g2l(A + (size_t)(m0 + r) * K + k0 + q * 8, &lA[buf][r * 32 + q * 8]);
    }
  };
  auto stageB = [&](int buf, int k0) {
#pragma unroll
    for (int idx = t; idx < 1024; idx += 128) {
      int r = idx >> 5, cu = idx & 31;              // row k, uint col (2 cols)
      unsigned w = *(const unsigned*)(W + (size_t)(k0 + r) * N + n0 + 2 * cu);
      lBt[buf][(2 * cu) * 32 + r]     = __builtin_bit_cast(__bf16, (unsigned short)(w & 0xffffu));
      lBt[buf][(2 * cu + 1) * 32 + r] = __builtin_bit_cast(__bf16, (unsigned short)(w >> 16));
    }
  };

  v8f acc0 = {}, acc1 = {}, acc2 = {}, acc3 = {};
  const int nk = K >> 5;

  stageA(0, 0);
  stageB(0, 0);
  async_wait0();
  __syncthreads();

  for (int ki = 0; ki < nk; ki++) {
    const int cur = ki & 1;
    if (ki + 1 < nk) {                     // stage next tile into other buffer
      stageA(cur ^ 1, (ki + 1) * 32);
      stageB(cur ^ 1, (ki + 1) * 32);
    }
    FragU af, b0, b1, b2, b3;
#pragma unroll
    for (int p = 0; p < 8; p++) {
      int kp = frag_k(p, half);            // always even -> b32 LDS loads
      af.u[p] = *(const unsigned*)(&lA[cur][(wr * 16 + lid) * 32 + kp]);
      b0.u[p] = *(const unsigned*)(&lBt[cur][(0 * 16 + lid) * 32 + kp]);
      b1.u[p] = *(const unsigned*)(&lBt[cur][(1 * 16 + lid) * 32 + kp]);
      b2.u[p] = *(const unsigned*)(&lBt[cur][(2 * 16 + lid) * 32 + kp]);
      b3.u[p] = *(const unsigned*)(&lBt[cur][(3 * 16 + lid) * 32 + kp]);
    }
    acc0 = __builtin_amdgcn_wmma_f32_16x16x32_bf16(false, af.v, false, b0.v, (short)0, acc0, false, false);
    acc1 = __builtin_amdgcn_wmma_f32_16x16x32_bf16(false, af.v, false, b1.v, (short)0, acc1, false, false);
    acc2 = __builtin_amdgcn_wmma_f32_16x16x32_bf16(false, af.v, false, b2.v, (short)0, acc2, false, false);
    acc3 = __builtin_amdgcn_wmma_f32_16x16x32_bf16(false, af.v, false, b3.v, (short)0, acc3, false, false);
    async_wait0();                         // next-tile async copies landed
    __syncthreads();
  }

  // ---- epilogue ----
  v8f accs[4] = {acc0, acc1, acc2, acc3};
#pragma unroll
  for (int tc = 0; tc < 4; tc++) {
    int n = n0 + tc * 16 + lid;
    float bval = (flags & 1) ? bias[n] : 0.0f;
#pragma unroll
    for (int r = 0; r < 8; r++) {
      int m = m0 + wr * 16 + r + 8 * half;
      float v = accs[tc][r] + bval;
      if (flags & 2) v = fmaxf(v, 0.0f);
      size_t idx = (size_t)m * N + n;
      if (flags & 4) Cb[idx] = (__bf16)v;
      else           Cf[idx] = v;
    }
  }
}

// ---------------- WMMA multi-head attention, one wave per (b, head) ----------------
__global__ void k_attn(const float* __restrict__ qkv, __bf16* __restrict__ o) {
  const int b = blockIdx.x >> 3, hh = blockIdx.x & 7;
  const int lane = threadIdx.x;
  const int half = lane >> 4, lid = lane & 15;
  __shared__ __align__(16) __bf16 Qb[112 * 16], Kb[112 * 16], Vb[112 * 16];
  __shared__ float  S[16 * 112];
  __shared__ __align__(16) __bf16 Pb[16 * 128];
  __shared__ float  rinv[16];
  for (int idx = lane; idx < 112 * 16; idx += 32) {
    int n = idx >> 4, k = idx & 15;
    float qv = 0.f, kv = 0.f, vv = 0.f;
    if (n < NN) {
      const float* base = qkv + (size_t)(b * NN + n) * 384 + hh * DK + k;
      qv = base[0] * 0.25f;     // fold 1/sqrt(DK) into Q
      kv = base[128];
      vv = base[256];
    }
    Qb[idx] = (__bf16)qv; Kb[idx] = (__bf16)kv; Vb[idx] = (__bf16)vv;
  }
  __syncthreads();
  for (int i = 0; i < 7; i++) {           // query row tiles
    FragU av;
#pragma unroll
    for (int p = 0; p < 8; p++) {
      int kp = frag_k(p, half);           // DK=16: upper half of K is zero-pad
      av.u[p] = (kp < DK) ? *(const unsigned*)(Qb + (i * 16 + lid) * 16 + kp) : 0u;
    }
    for (int j = 0; j < 7; j++) {         // key col tiles: S = Q @ K^T
      FragU bv;
#pragma unroll
      for (int p = 0; p < 8; p++) {
        int kp = frag_k(p, half);
        bv.u[p] = (kp < DK) ? *(const unsigned*)(Kb + (j * 16 + lid) * 16 + kp) : 0u;
      }
      v8f c = {};
      c = __builtin_amdgcn_wmma_f32_16x16x32_bf16(false, av.v, false, bv.v,
                                                  (short)0, c, false, false);
#pragma unroll
      for (int r = 0; r < 8; r++) S[(r + 8 * half) * 112 + j * 16 + lid] = c[r];
    }
    __syncthreads();
    if (lane < 16) {                       // row softmax over valid keys
      float mx = -1e30f;
      for (int n = 0; n < NN; n++) mx = fmaxf(mx, S[lane * 112 + n]);
      float sum = 0.f;
      for (int n = 0; n < NN; n++) { float e = __expf(S[lane * 112 + n] - mx); S[lane * 112 + n] = e; sum += e; }
      rinv[lane] = 1.0f / sum;
    }
    __syncthreads();
    for (int idx = lane; idx < 16 * 128; idx += 32) {
      int r = idx >> 7, cc = idx & 127;
      Pb[idx] = (cc < NN) ? (__bf16)(S[r * 112 + cc] * rinv[r]) : (__bf16)0.0f;
    }
    __syncthreads();
    v8f oacc = {};
    for (int ks = 0; ks < 4; ks++) {       // O = P @ V, K padded 112->128
      int kb = ks * 32;
      FragU pa, vb;
#pragma unroll
      for (int p = 0; p < 8; p++) {
        int kp = kb + frag_k(p, half);
        pa.u[p] = *(const unsigned*)(Pb + lid * 128 + kp);
        unsigned lo = (kp < 112)     ? (unsigned)__builtin_bit_cast(unsigned short, Vb[kp * 16 + lid])       : 0u;
        unsigned hi = (kp + 1 < 112) ? (unsigned)__builtin_bit_cast(unsigned short, Vb[(kp + 1) * 16 + lid]) : 0u;
        vb.u[p] = lo | (hi << 16);
      }
      oacc = __builtin_amdgcn_wmma_f32_16x16x32_bf16(false, pa.v, false, vb.v,
                                                     (short)0, oacc, false, false);
    }
#pragma unroll
    for (int r = 0; r < 8; r++) {
      int n = i * 16 + r + 8 * half;
      if (n < NN) o[(size_t)(b * NN + n) * DD + hh * DK + lid] = (__bf16)oacc[r];
    }
    __syncthreads();
  }
}

// ---------------- BatchNorm (training mode) over B*N per channel ----------------
__global__ void k_bnstats(const float* __restrict__ h, const float* __restrict__ o,
                          float* __restrict__ mv) {
  int c = blockIdx.x, t = threadIdx.x;     // 128 blocks x 256 threads
  __shared__ float ssum[256], ssq[256];
  float s = 0.f, q = 0.f;
  for (int i = t; i < MROWS; i += 256) {
    float v = h[(size_t)i * DD + c] + o[(size_t)i * DD + c];
    s += v; q += v * v;
  }
  ssum[t] = s; ssq[t] = q; __syncthreads();
  for (int w = 128; w > 0; w >>= 1) {
    if (t < w) { ssum[t] += ssum[t + w]; ssq[t] += ssq[t + w]; }
    __syncthreads();
  }
  if (t == 0) {
    float m = ssum[0] / (float)MROWS;
    mv[c] = m;
    mv[DD + c] = ssq[0] / (float)MROWS - m * m;
  }
}

__global__ void k_bnapply(float* __restrict__ h, const float* __restrict__ o,
                          const float* __restrict__ mv, const float* __restrict__ gb,
                          __bf16* __restrict__ hb) {
  size_t i = (size_t)blockIdx.x * 256 + threadIdx.x;   // MROWS*DD
  int c = (int)(i & (DD - 1));
  float x = h[i] + o[i];
  float y = gb[c] * (x - mv[c]) * rsqrtf(mv[DD + c] + 1e-5f) + gb[DD + c];
  h[i] = y;
  hb[i] = (__bf16)y;
}

// ---------------- graph mean over N -> bf16 ----------------
__global__ void k_mean(const float* __restrict__ h, __bf16* __restrict__ mb) {
  int b = blockIdx.x, d = threadIdx.x;     // 512 x 128
  float s = 0.f;
  for (int n = 0; n < NN; n++) s += h[((size_t)b * NN + n) * DD + d];
  mb[b * DD + d] = (__bf16)(s * (1.0f / NN));
}

// ---------------- decode state reset ----------------
__global__ void k_reset(int* __restrict__ mask, float* __restrict__ ll,
                        int* __restrict__ prevA, int* __restrict__ firstA) {
  int i = blockIdx.x * 256 + threadIdx.x;
  if (i < BB * NN) mask[i] = 0;
  if (i < BB) { ll[i] = 0.f; prevA[i] = 0; firstA[i] = 0; }
}

// ---------------- one greedy pointer-decode step, one block per batch elem ----------------
__global__ void k_decode(int step, const float* __restrict__ h,
                         const float* __restrict__ placeholder,
                         const float* __restrict__ fixed_ctx,
                         const float* __restrict__ stepW,
                         const float* __restrict__ outW,
                         const float* __restrict__ gKVL,
                         int* __restrict__ mask, int* __restrict__ pi,
                         float* __restrict__ ll, int* __restrict__ prevA,
                         int* __restrict__ firstA) {
  const int b = blockIdx.x, t = threadIdx.x;   // 512 x 128
  __shared__ float ctx[256], q[128], sc[800], gl[128], g2[128], lg[100], rtmp[8];
  const float nf = 0.08838834764831845f;       // 1/sqrt(128)

  if (step == 0) {
    ctx[t] = placeholder[t];
    ctx[t + 128] = placeholder[t + 128];
  } else {
    int fa = firstA[b], pa = prevA[b];
    ctx[t]       = h[((size_t)b * NN + fa) * DD + t];
    ctx[t + 128] = h[((size_t)b * NN + pa) * DD + t];
  }
  __syncthreads();
  {
    float s = fixed_ctx[(size_t)b * DD + t];
#pragma unroll 8
    for (int k = 0; k < 256; k++) s += ctx[k] * stepW[k * DD + t];
    q[t] = s;
  }
  __syncthreads();
  for (int p = t; p < 800; p += 128) {          // glimpse compatibilities
    int hh = p / NN, n = p % NN;
    const float* gk = gKVL + (size_t)(b * NN + n) * 384 + hh * DK;
    float s = 0.f;
#pragma unroll
    for (int k = 0; k < DK; k++) s += q[hh * DK + k] * gk[k];
    s *= nf;
    if (mask[b * NN + n]) s = NEGI;
    sc[p] = s;
  }
  __syncthreads();
  if (t < 8) {                                   // per-head softmax
    float mx = -1e30f;
    for (int n = 0; n < NN; n++) mx = fmaxf(mx, sc[t * NN + n]);
    float sum = 0.f;
    for (int n = 0; n < NN; n++) { float e = __expf(sc[t * NN + n] - mx); sc[t * NN + n] = e; sum += e; }
    rtmp[t] = 1.0f / sum;
  }
  __syncthreads();
  {
    int hh = t / DK, k = t % DK;
    float s = 0.f;
    for (int n = 0; n < NN; n++)
      s += sc[hh * NN + n] * gKVL[(size_t)(b * NN + n) * 384 + 128 + hh * DK + k];
    gl[t] = s * rtmp[hh];
  }
  __syncthreads();
  {
    float s = 0.f;
#pragma unroll 8
    for (int k = 0; k < DD; k++) s += gl[k] * outW[k * DD + t];
    g2[t] = s;
  }
  __syncthreads();
  if (t < NN) {                                  // pointer logits
    const float* lk = gKVL + (size_t)(b * NN + t) * 384 + 256;
    float s = 0.f;
#pragma unroll 8
    for (int d = 0; d < DD; d++) s += g2[d] * lk[d];
    s = tanhf(s * nf) * 10.0f;
    if (mask[b * NN + t]) s = NEGI;
    lg[t] = s;
  }
  __syncthreads();
  if (t == 0) {                                  // greedy select + log-prob
    float mx = -1e30f; int sel = 0;
    for (int n = 0; n < NN; n++) if (lg[n] > mx) { mx = lg[n]; sel = n; }
    float sum = 0.f;
    for (int n = 0; n < NN; n++) sum += __expf(lg[n] - mx);
    float logZ = mx + __logf(sum);
    ll[b] += lg[sel] - logZ;
    pi[b * NN + step] = sel;
    mask[b * NN + sel] = 1;
    prevA[b] = sel;
    if (step == 0) firstA[b] = sel;
  }
}

// ---------------- tour cost + output packing ----------------
__global__ void k_final(const float* __restrict__ x, const int* __restrict__ pi,
                        const float* __restrict__ ll, float* __restrict__ out) {
  int b = blockIdx.x, t = threadIdx.x;           // 512 x 128
  if (t == 0) {
    float cost = 0.f;
    int p0 = pi[b * NN];
    int prev = p0;
    for (int s = 1; s < NN; s++) {
      int cur = pi[b * NN + s];
      float dx = x[(size_t)(b * NN + cur) * 2]     - x[(size_t)(b * NN + prev) * 2];
      float dy = x[(size_t)(b * NN + cur) * 2 + 1] - x[(size_t)(b * NN + prev) * 2 + 1];
      cost += sqrtf(dx * dx + dy * dy);
      prev = cur;
    }
    float dx = x[(size_t)(b * NN + p0) * 2]     - x[(size_t)(b * NN + prev) * 2];
    float dy = x[(size_t)(b * NN + p0) * 2 + 1] - x[(size_t)(b * NN + prev) * 2 + 1];
    cost += sqrtf(dx * dx + dy * dy);
    out[b] = cost;
    out[BB + b] = ll[b];
  }
  for (int s = t; s < NN; s += 128)
    out[2 * BB + b * NN + s] = (float)pi[b * NN + s];
}

// ================= host side =================
extern "C" void kernel_launch(void* const* d_in, const int* in_sizes, int n_in,
                              void* d_out, int out_size, void* d_ws, size_t ws_size,
                              hipStream_t stream) {
  const float* x      = (const float*)d_in[0];
  const float* initW  = (const float*)d_in[1];
  const float* initb  = (const float*)d_in[2];
  const float* qkvW   = (const float*)d_in[3];   // [2,128,384]
  const float* outW   = (const float*)d_in[4];   // [2,128,128]
  const float* bn1    = (const float*)d_in[5];   // [2,2,128]
  const float* bn2    = (const float*)d_in[6];
  const float* ff1W   = (const float*)d_in[7];   // [2,128,512]
  const float* ff1b   = (const float*)d_in[8];   // [2,512]
  const float* ff2W   = (const float*)d_in[9];   // [2,512,128]
  const float* ff2b   = (const float*)d_in[10];  // [2,128]
  const float* wph    = (const float*)d_in[11];  // [256]
  const float* nodesW = (const float*)d_in[12];  // [128,384]
  const float* fixedW = (const float*)d_in[13];  // [128,128]
  const float* stepW  = (const float*)d_in[14];  // [256,128]
  const float* poutW  = (const float*)d_in[15];  // [128,128]
  float* out = (float*)d_out;

  // ---- workspace layout (bytes) ----
  char* ws = (char*)d_ws;
  size_t off = 0;
  auto alloc = [&](size_t bytes) { char* p = ws + off; off = (off + bytes + 255) & ~(size_t)255; return p; };
  __bf16* hb    = (__bf16*)alloc((size_t)MROWS * DD * 2);
  float*  h     = (float*) alloc((size_t)MROWS * DD * 4);
  float*  tmp1  = (float*) alloc((size_t)MROWS * 384 * 4);   // qkv / gKVL
  __bf16* tmpb  = (__bf16*)alloc((size_t)MROWS * FF_ * 2);   // attn-out / ff1-out
  float*  tmp2  = (float*) alloc((size_t)MROWS * DD * 4);
  __bf16* wb    = (__bf16*)alloc(458752 * 2);
  __bf16* meanb = (__bf16*)alloc((size_t)BB * DD * 2);
  float*  fixc  = (float*) alloc((size_t)BB * DD * 4);
  float*  mv    = (float*) alloc(2 * DD * 4);
  int*    mask  = (int*)   alloc((size_t)BB * NN * 4);
  int*    pi    = (int*)   alloc((size_t)BB * NN * 4);
  float*  ll    = (float*) alloc(BB * 4);
  int*    prevA = (int*)   alloc(BB * 4);
  int*    firstA= (int*)   alloc(BB * 4);
  (void)ws_size; (void)n_in; (void)in_sizes; (void)out_size;

  __bf16* wb_qkv   = wb;             // 2*128*384 = 98304
  __bf16* wb_out   = wb + 98304;     // 2*128*128 = 32768
  __bf16* wb_ff1   = wb + 131072;    // 2*128*512 = 131072
  __bf16* wb_ff2   = wb + 262144;    // 2*512*128 = 131072
  __bf16* wb_nodes = wb + 393216;    // 128*384  = 49152
  __bf16* wb_fixed = wb + 442368;    // 128*128  = 16384

  auto cvt = [&](const float* s, __bf16* d, int n) {
    hipLaunchKernelGGL(k_f2bf, dim3((n + 255) / 256), dim3(256), 0, stream, s, d, n);
  };
  cvt(qkvW,   wb_qkv,   98304);
  cvt(outW,   wb_out,   32768);
  cvt(ff1W,   wb_ff1,   131072);
  cvt(ff2W,   wb_ff2,   131072);
  cvt(nodesW, wb_nodes, 49152);
  cvt(fixedW, wb_fixed, 16384);

  // init projection
  hipLaunchKernelGGL(k_init, dim3(MROWS * DD / 256), dim3(256), 0, stream,
                     x, initW, initb, h, hb);

  const int MB64 = MROWS / 64;  // 800 row blocks
  for (int l = 0; l < 2; l++) {
    // QKV
    hipLaunchKernelGGL(k_gemm, dim3(384 / 64, MB64), dim3(128), 0, stream,
                       hb, wb_qkv + l * 49152, (const float*)nullptr,
                       tmp1, tmpb, MROWS, DD, 384, 0);
    // attention -> bf16
    hipLaunchKernelGGL(k_attn, dim3(BB * HH), dim3(32), 0, stream, tmp1, tmpb);
    // out projection -> f32 tmp2
    hipLaunchKernelGGL(k_gemm, dim3(DD / 64, MB64), dim3(128), 0, stream,
                       tmpb, wb_out + l * 16384, (const float*)nullptr,
                       tmp2, tmpb, MROWS, DD, DD, 0);
    // BN1(h + o)
    hipLaunchKernelGGL(k_bnstats, dim3(DD), dim3(256), 0, stream, h, tmp2, mv);
    hipLaunchKernelGGL(k_bnapply, dim3(MROWS * DD / 256), dim3(256), 0, stream,
                       h, tmp2, mv, bn1 + l * 256, hb);
    // FF1 (bias+relu, bf16 out)
    hipLaunchKernelGGL(k_gemm, dim3(FF_ / 64, MB64), dim3(128), 0, stream,
                       hb, wb_ff1 + l * 65536, ff1b + l * FF_,
                       tmp2, tmpb, MROWS, DD, FF_, 1 | 2 | 4);
    // FF2 (bias, f32 out)
    hipLaunchKernelGGL(k_gemm, dim3(DD / 64, MB64), dim3(128), 0, stream,
                       tmpb, wb_ff2 + l * 65536, ff2b + l * DD,
                       tmp2, tmpb, MROWS, FF_, DD, 1);
    // BN2(h + ff)
    hipLaunchKernelGGL(k_bnstats, dim3(DD), dim3(256), 0, stream, h, tmp2, mv);
    hipLaunchKernelGGL(k_bnapply, dim3(MROWS * DD / 256), dim3(256), 0, stream,
                       h, tmp2, mv, bn2 + l * 256, hb);
  }

  // decoder precompute
  hipLaunchKernelGGL(k_mean, dim3(BB), dim3(DD), 0, stream, h, meanb);
  hipLaunchKernelGGL(k_gemm, dim3(DD / 64, BB / 64), dim3(128), 0, stream,
                     meanb, wb_fixed, (const float*)nullptr,
                     fixc, tmpb, BB, DD, DD, 0);
  hipLaunchKernelGGL(k_gemm, dim3(384 / 64, MB64), dim3(128), 0, stream,
                     hb, wb_nodes, (const float*)nullptr,
                     tmp1, tmpb, MROWS, DD, 384, 0);   // gK|gV|lK

  // greedy autoregressive decode
  hipLaunchKernelGGL(k_reset, dim3((BB * NN + 255) / 256), dim3(256), 0, stream,
                     mask, ll, prevA, firstA);
  for (int s = 0; s < NN; s++) {
    hipLaunchKernelGGL(k_decode, dim3(BB), dim3(128), 0, stream,
                       s, h, wph, fixc, stepW, poutW, tmp1,
                       mask, pi, ll, prevA, firstA);
  }
  hipLaunchKernelGGL(k_final, dim3(BB), dim3(128), 0, stream, x, pi, ll, out);
}
